// MultiTimescaleHybridRecurrentFFN_69123203662005
// MI455X (gfx1250) — matrix operand order repaired and checked by reading, not compiled, not run on hardware
//
#include <hip/hip_runtime.h>
#include <cstdint>

// Problem constants (fixed by the reference)
#define NSC 4
#define CHN 256
#define AA  1024          // NS*CH
#define HH  1024
#define LL  1024
#define BBB 4
#define SSS 4096
#define TT  (BBB*SSS)     // 16384 tokens

// ---------- bf16 helpers (header-independent) ----------
typedef __attribute__((ext_vector_type(16))) __bf16 bf16x16;
typedef __attribute__((ext_vector_type(8)))  float  floatx8;

__device__ __forceinline__ unsigned short f32_to_bf16(float f) {
  union { float f; unsigned int u; } c; c.f = f;
  unsigned int u = c.u;
  u += 0x7fffu + ((u >> 16) & 1u);   // round-to-nearest-even
  return (unsigned short)(u >> 16);
}
__device__ __forceinline__ float bf16_to_f32(unsigned short h) {
  union { unsigned int u; float f; } c; c.u = ((unsigned int)h) << 16;
  return c.f;
}
__device__ __forceinline__ float sigmoidf_(float x) { return 1.f / (1.f + __expf(-x)); }
__device__ __forceinline__ float siluf_(float x)    { return x / (1.f + __expf(-x)); }

// ---------- CDNA5 async global->LDS DMA (ASYNCcnt-tracked, no VGPR round trip) ----------
__device__ __forceinline__ void async_g2l_b128(const void* gptr, void* ldsptr) {
  uint32_t l = (uint32_t)(uintptr_t)ldsptr;      // LDS aperture: addr[31:0] is the LDS address
  uint64_t g = (uint64_t)(uintptr_t)gptr;
  asm volatile("global_load_async_to_lds_b128 %0, %1, off"
               :: "v"(l), "v"(g) : "memory");
}
__device__ __forceinline__ void wait_async0() {
  asm volatile("s_wait_asynccnt 0" ::: "memory");
}

// ---------- fp32 -> bf16 convert ----------
__global__ void k_cvt_bf16(const float* __restrict__ src,
                           unsigned short* __restrict__ dst, int n) {
  int i = blockIdx.x * blockDim.x + threadIdx.x;
  if (i < n) dst[i] = f32_to_bf16(src[i]);
}

// ---------- WMMA GEMM: C[T,N] = Xb[T,K] * Wb[N,K]^T (Linear-style) ----------
// Block tile 256x128, 8 waves in 4x2 grid; each wave computes 64x64 = 4x4 WMMA tiles.
// Per 32-K step a wave does 16 ds_load_b128 for 16 WMMAs -> LDS-read clocks ~= XDL clocks.
#define TM   256
#define TN   128
#define TKK  32
#define APAD 40   // elements per As row (80B: keeps 16B alignment, spreads banks)
#define BPAD 40

__global__ __launch_bounds__(256, 1)
void k_gemm_bf16(const unsigned short* __restrict__ Xb,
                 const unsigned short* __restrict__ Wb,
                 float* __restrict__ Cf,
                 unsigned short* __restrict__ Cb,
                 int N, int K, int outBf16, int accumulate)
{
  __shared__ unsigned short As[2][TM * APAD];   // 2 x 256 x 40 x 2B = 40 KB
  __shared__ unsigned short Bs[2][TN * BPAD];   // 2 x 128 x 40 x 2B = 20 KB

  const int tid  = threadIdx.x;
  const int lane = tid & 31;
  const int wave = tid >> 5;     // 0..7
  const int wm   = wave >> 1;    // 0..3 -> 64-row group
  const int wn   = wave & 1;     // 0..1 -> 64-col group
  const int half = lane >> 4;
  const int l16  = lane & 15;

  const int mBase = blockIdx.y * TM;
  const int nBase = blockIdx.x * TN;

  floatx8 acc[4][4];
#pragma unroll
  for (int r = 0; r < 4; ++r)
#pragma unroll
    for (int c = 0; c < 4; ++c) acc[r][c] = (floatx8)0.0f;

  const int nk = K / TKK;

  auto stage = [&](int buf, int k0) {
    // X tile: 256 rows x 32 bf16 -> 1024 x 16B chunks, 4 per thread
#pragma unroll
    for (int i = 0; i < 4; ++i) {
      int lin = tid + i * 256;
      int row = lin >> 2;
      int col = (lin & 3) * 8;
      async_g2l_b128(Xb + (size_t)(mBase + row) * K + k0 + col,
                     &As[buf][row * APAD + col]);
    }
    // W tile: 128 rows x 32 bf16 -> 512 x 16B chunks, 2 per thread
#pragma unroll
    for (int i = 0; i < 2; ++i) {
      int lin = tid + i * 256;
      int row = lin >> 2;
      int col = (lin & 3) * 8;
      async_g2l_b128(Wb + (size_t)(nBase + row) * K + k0 + col,
                     &Bs[buf][row * BPAD + col]);
    }
  };

  stage(0, 0);
  wait_async0();
  __syncthreads();

  for (int kt = 0; kt < nk; ++kt) {
    const int cur = kt & 1;
    if (kt + 1 < nk) stage(cur ^ 1, (kt + 1) * TKK);

    // B fragments: 32x16 bf16 layout: lane n=l16; lanes 0-15 K=0..7 & 16..23 ; lanes 16-31 K=8..15 & 24..31
    union { bf16x16 v; uint4 q[2]; } bfr[4];
#pragma unroll
    for (int c = 0; c < 4; ++c) {
      int col = wn * 64 + c * 16 + l16;
      int off = col * BPAD + half * 8;
      bfr[c].q[0] = *(const uint4*)(&Bs[cur][off]);
      bfr[c].q[1] = *(const uint4*)(&Bs[cur][off + 16]);
    }
    // A fragments: 16x32 bf16 layout: lanes 0-15 K=0..7 & 16..23 ; lanes 16-31 K=8..15 & 24..31
#pragma unroll
    for (int r = 0; r < 4; ++r) {
      union { bf16x16 v; uint4 q[2]; } afr;
      int row = wm * 64 + r * 16 + l16;
      int off = row * APAD + half * 8;
      afr.q[0] = *(const uint4*)(&As[cur][off]);
      afr.q[1] = *(const uint4*)(&As[cur][off + 16]);
#pragma unroll
      for (int c = 0; c < 4; ++c)
        acc[r][c] = __builtin_amdgcn_wmma_f32_16x16x32_bf16(
            false, afr.v, false, bfr[c].v, (short)0, acc[r][c], false, false);
    }

    // own async DMAs landed, then whole workgroup synced -> next buffer valid
    wait_async0();
    __syncthreads();
  }

  // Epilogue. C layout: VGPR i -> M = i + 8*half ; lane l16 -> N
#pragma unroll
  for (int r = 0; r < 4; ++r) {
#pragma unroll
    for (int c = 0; c < 4; ++c) {
      int col = nBase + wn * 64 + c * 16 + l16;
#pragma unroll
      for (int i = 0; i < 8; ++i) {
        int row = mBase + wm * 64 + r * 16 + half * 8 + i;
        size_t idx = (size_t)row * N + col;
        float v = acc[r][c][i];
        if (outBf16) {
          Cb[idx] = f32_to_bf16(v);
        } else if (accumulate) {
          Cf[idx] = Cf[idx] + v;
        } else {
          Cf[idx] = v;
        }
      }
    }
  }
}

// ---------- mix logits + softmax: one wave per token ----------
__global__ void k_mix_softmax(const float* __restrict__ x,
                              const float* __restrict__ Wmix,
                              float* __restrict__ smix)
{
  int wave  = threadIdx.x >> 5;
  int lane  = threadIdx.x & 31;
  int token = blockIdx.x * (blockDim.x >> 5) + wave;
  if (token >= TT) return;
  const float* xr = x + (size_t)token * HH;
  float a0 = 0.f, a1 = 0.f, a2 = 0.f, a3 = 0.f;
  for (int k = lane; k < HH; k += 32) {
    float xv = xr[k];
    a0 = fmaf(xv, Wmix[0 * HH + k], a0);
    a1 = fmaf(xv, Wmix[1 * HH + k], a1);
    a2 = fmaf(xv, Wmix[2 * HH + k], a2);
    a3 = fmaf(xv, Wmix[3 * HH + k], a3);
  }
#pragma unroll
  for (int off = 16; off > 0; off >>= 1) {
    a0 += __shfl_down(a0, off, 32);
    a1 += __shfl_down(a1, off, 32);
    a2 += __shfl_down(a2, off, 32);
    a3 += __shfl_down(a3, off, 32);
  }
  if (lane == 0) {
    float m = fmaxf(fmaxf(a0, a1), fmaxf(a2, a3));
    float e0 = __expf(a0 - m), e1 = __expf(a1 - m), e2 = __expf(a2 - m), e3 = __expf(a3 - m);
    float inv = 1.f / (e0 + e1 + e2 + e3);
    float* o = smix + (size_t)token * 4;
    o[0] = e0 * inv; o[1] = e1 * inv; o[2] = e2 * inv; o[3] = e3 * inv;
  }
}

// ---------- update = sigmoid(pin) * silu(pval) ----------
__global__ void k_update(const unsigned short* __restrict__ pin,
                         const unsigned short* __restrict__ pval,
                         float* __restrict__ upd, int n)
{
  int i = blockIdx.x * blockDim.x + threadIdx.x;
  if (i < n) {
    float g = sigmoidf_(bf16_to_f32(pin[i]));
    float c = siluf_(bf16_to_f32(pval[i]));
    upd[i] = g * c;
  }
}

// ---------- linear-decay scan, in place: upd -> states ----------
__global__ void k_scan(float* __restrict__ st,
                       const float* __restrict__ logit_decay,
                       const float* __restrict__ init_state)
{
  int ch = blockIdx.x * blockDim.x + threadIdx.x;  // 0..B*A-1
  if (ch >= BBB * AA) return;
  int b = ch / AA, a = ch % AA;
  float d = sigmoidf_(logit_decay[a]);
  float s = init_state[a];
  float* p = st + (size_t)b * SSS * AA + a;
#pragma unroll 4
  for (int t = 0; t < SSS; ++t) {
    float u = *p;
    s = fmaf(d, s, u);
    *p = s;
    p += AA;
  }
}

// ---------- read = silu(q * state) * scale_mix -> bf16 ----------
__global__ void k_read(const unsigned short* __restrict__ pq,
                       const float* __restrict__ states,
                       const float* __restrict__ smix,
                       unsigned short* __restrict__ readb, int n)
{
  int i = blockIdx.x * blockDim.x + threadIdx.x;
  if (i < n) {
    int a = i & (AA - 1);
    int t = i >> 10;
    float q = bf16_to_f32(pq[i]);
    float v = q * states[i];
    float r = siluf_(v) * smix[t * 4 + (a >> 8)];
    readb[i] = f32_to_bf16(r);
  }
}

// ---------- SwiGLU hidden = up * silu(gate) -> bf16 ----------
__global__ void k_hidden(const unsigned short* __restrict__ pup,
                         const unsigned short* __restrict__ pgate,
                         unsigned short* __restrict__ hb, int n)
{
  int i = blockIdx.x * blockDim.x + threadIdx.x;
  if (i < n) {
    float u = bf16_to_f32(pup[i]);
    float g = siluf_(bf16_to_f32(pgate[i]));
    hb[i] = f32_to_bf16(u * g);
  }
}

extern "C" void kernel_launch(void* const* d_in, const int* in_sizes, int n_in,
                              void* d_out, int out_size, void* d_ws, size_t ws_size,
                              hipStream_t stream) {
  const float* x       = (const float*)d_in[0];
  const float* W_in    = (const float*)d_in[1];
  const float* W_val   = (const float*)d_in[2];
  const float* W_q     = (const float*)d_in[3];
  const float* W_mix   = (const float*)d_in[4];
  const float* W_rout  = (const float*)d_in[5];
  const float* W_up    = (const float*)d_in[6];
  const float* W_gate  = (const float*)d_in[7];
  const float* W_down  = (const float*)d_in[8];
  const float* ldecay  = (const float*)d_in[9];
  const float* istate  = (const float*)d_in[10];
  float* out = (float*)d_out;

  // workspace carve-up
  char* w = (char*)d_ws;
  auto alloc = [&](size_t bytes) -> void* {
    void* p = (void*)w;
    w += (bytes + 255) & ~(size_t)255;
    return p;
  };
  const size_t TA = (size_t)TT * AA;           // 16.7M elements
  unsigned short* xb     = (unsigned short*)alloc((size_t)TT * HH * 2);
  unsigned short* wbin   = (unsigned short*)alloc((size_t)AA * HH * 2);
  unsigned short* wbval  = (unsigned short*)alloc((size_t)AA * HH * 2);
  unsigned short* wbq    = (unsigned short*)alloc((size_t)AA * HH * 2);
  unsigned short* wbup   = (unsigned short*)alloc((size_t)LL * HH * 2);
  unsigned short* wbgate = (unsigned short*)alloc((size_t)LL * HH * 2);
  unsigned short* wbrout = (unsigned short*)alloc((size_t)HH * AA * 2);
  unsigned short* wbdown = (unsigned short*)alloc((size_t)HH * LL * 2);
  unsigned short* pin    = (unsigned short*)alloc(TA * 2);
  unsigned short* pval   = (unsigned short*)alloc(TA * 2);
  unsigned short* pq     = (unsigned short*)alloc(TA * 2);
  unsigned short* pup    = (unsigned short*)alloc(TA * 2);
  unsigned short* pgate  = (unsigned short*)alloc(TA * 2);
  float*          upd    = (float*)alloc(TA * 4);          // becomes states in place
  unsigned short* readb  = (unsigned short*)alloc(TA * 2);
  unsigned short* hb     = (unsigned short*)alloc(TA * 2);
  float*          smix   = (float*)alloc((size_t)TT * 4 * 4);

  const int nTok  = TT * HH;       // 16,777,216
  const int nW    = AA * HH;       // 1,048,576
  dim3 blk(256);

  // 1) fp32 -> bf16 conversions
  k_cvt_bf16<<<dim3((nTok + 255) / 256), blk, 0, stream>>>(x, xb, nTok);
  k_cvt_bf16<<<dim3((nW + 255) / 256), blk, 0, stream>>>(W_in,   wbin,   nW);
  k_cvt_bf16<<<dim3((nW + 255) / 256), blk, 0, stream>>>(W_val,  wbval,  nW);
  k_cvt_bf16<<<dim3((nW + 255) / 256), blk, 0, stream>>>(W_q,    wbq,    nW);
  k_cvt_bf16<<<dim3((nW + 255) / 256), blk, 0, stream>>>(W_up,   wbup,   nW);
  k_cvt_bf16<<<dim3((nW + 255) / 256), blk, 0, stream>>>(W_gate, wbgate, nW);
  k_cvt_bf16<<<dim3((nW + 255) / 256), blk, 0, stream>>>(W_rout, wbrout, nW);
  k_cvt_bf16<<<dim3((nW + 255) / 256), blk, 0, stream>>>(W_down, wbdown, nW);

  // 2) token projections via WMMA GEMM (bf16 out)
  dim3 ggrid(AA / TN, TT / TM);   // (8, 64)
  k_gemm_bf16<<<ggrid, blk, 0, stream>>>(xb, wbin,   nullptr, pin,   AA, HH, 1, 0);
  k_gemm_bf16<<<ggrid, blk, 0, stream>>>(xb, wbval,  nullptr, pval,  AA, HH, 1, 0);
  k_gemm_bf16<<<ggrid, blk, 0, stream>>>(xb, wbq,    nullptr, pq,    AA, HH, 1, 0);
  k_gemm_bf16<<<ggrid, blk, 0, stream>>>(xb, wbup,   nullptr, pup,   LL, HH, 1, 0);
  k_gemm_bf16<<<ggrid, blk, 0, stream>>>(xb, wbgate, nullptr, pgate, LL, HH, 1, 0);

  // 3) mix softmax (one wave per token; 4 waves per block)
  k_mix_softmax<<<dim3(TT / 4), dim3(128), 0, stream>>>(x, W_mix, smix);

  // 4) update, scan (in place), read
  k_update<<<dim3((nTok + 255) / 256), blk, 0, stream>>>(pin, pval, upd, nTok);
  k_scan<<<dim3((BBB * AA + 255) / 256), blk, 0, stream>>>(upd, ldecay, istate);
  k_read<<<dim3((nTok + 255) / 256), blk, 0, stream>>>(pq, upd, smix, readb, nTok);

  // 5) rec_out = read @ W_rout^T  (f32 out, beta=0)
  k_gemm_bf16<<<ggrid, blk, 0, stream>>>(readb, wbrout, out, nullptr, HH, AA, 0, 0);

  // 6) SwiGLU local branch, accumulate into out (beta=1)
  k_hidden<<<dim3((nTok + 255) / 256), blk, 0, stream>>>(pup, pgate, hb, nTok);
  k_gemm_bf16<<<ggrid, blk, 0, stream>>>(hb, wbdown, out, nullptr, HH, LL, 0, 1);

  (void)in_sizes; (void)n_in; (void)out_size; (void)ws_size;
}
